// MHSA_8340826489355
// MI455X (gfx1250) — matrix operand order repaired and checked by reading, not compiled, not run on hardware
//
#include <hip/hip_runtime.h>
#include <hip/hip_bf16.h>

#define T  2048
#define D  2048
#define NH 16
#define HD 128

typedef __attribute__((ext_vector_type(16))) __bf16 v16bf;
typedef __attribute__((ext_vector_type(8)))  __bf16 v8bf;
typedef __attribute__((ext_vector_type(8)))  float  v8f;

union Frag16 { v16bf v; v8bf h[2]; };

// ---- CDNA5 async global->LDS copies (ASYNCcnt path), per CDNA5 ISA ----
__device__ inline void async_g2l_b64(unsigned lds_byte, const void* g) {
  asm volatile("global_load_async_to_lds_b64 %0, %1, off"
               :: "v"(lds_byte), "v"(g) : "memory");
}
__device__ inline void async_g2l_b128(unsigned lds_byte, const void* g) {
  asm volatile("global_load_async_to_lds_b128 %0, %1, off"
               :: "v"(lds_byte), "v"(g) : "memory");
}
__device__ inline void wait_async0() {
  asm volatile("s_wait_asynccnt 0" ::: "memory");
}
__device__ inline unsigned lds_addr32(const void* p) {
  // generic shared pointer: low 32 bits are the LDS byte offset
  return (unsigned)(unsigned long long)p;
}

// A-matrix fragment (16x32 bf16), row-major source with leading dim `ld`.
// Contiguous k order used here (same as B): lane L holds row M=L%16; lanes<16: K=0..15, lanes>=16: K=16..31.
__device__ inline v16bf load_a_frag(const __bf16* base, int ld) {
  int lane = threadIdx.x & 31;
  int row  = lane & 15;
  int klo  = (lane < 16) ? 0 : 16;
  const __bf16* p = base + (long)row * ld + klo;
  Frag16 f;
  f.h[0] = *(const v8bf*)(p);
  f.h[1] = *(const v8bf*)(p + 8);
  return f.v;
}

// B-matrix fragment (32x16 bf16) from a transposed (N x K row-major) source:
// lane L holds col N=L%16; lanes<16: K=0..15, lanes>=16: K=16..31 (2 per VGPR).
__device__ inline v16bf load_b_frag(const __bf16* bt, int ld, int k0) {
  int lane = threadIdx.x & 31;
  int n    = lane & 15;
  int kb   = (lane < 16) ? 0 : 16;
  const __bf16* p = bt + (long)n * ld + k0 + kb;
  Frag16 f;
  f.h[0] = *(const v8bf*)(p);
  f.h[1] = *(const v8bf*)(p + 8);
  return f.v;
}

__global__ void cvt_f32_bf16(const float* __restrict__ s, __bf16* __restrict__ d, int n) {
  int i = blockIdx.x * blockDim.x + threadIdx.x;
  if (i < n) d[i] = (__bf16)s[i];
}

// (R x C) f32 row-major -> (C x R) bf16 row-major, batched along blockIdx.z
__global__ void transpose_f32_bf16(const float* __restrict__ s, __bf16* __restrict__ d,
                                   int R, int C) {
  long base = (long)blockIdx.z * R * C;
  int i = blockIdx.x * blockDim.x + threadIdx.x;
  if (i < R * C) {
    int r = i / C, c = i % C;
    d[base + (long)c * R + r] = (__bf16)s[base + i];
  }
}

// C(M x N) = A(M x K, bf16 rm) * B(K x N) with B given transposed (N x K rm).
// 8 waves/block; wave w owns the 16x16 tile at column blockIdx.y*128 + 16w.
// A tile (16x64) is async-staged to LDS (double buffered) and shared by all 8 waves;
// B fragments stream straight from global (contiguous b128 pairs, pipelined vs WMMA).
// mode 0: bf16 rm out; mode 1: bf16 transposed out (ldc = M); mode 2: f32 rm out.
__global__ __launch_bounds__(256) void gemm_bf16_wmma(
    const __bf16* __restrict__ A, const __bf16* __restrict__ Bt, void* __restrict__ Cv,
    int M, int N, int K, int ldc, int mode,
    long strideA, long strideB, long strideC) {
  __shared__ __bf16 sA[2][16 * 64];  // 2 x 2KB double buffer

  int batch = blockIdx.z;
  const __bf16* a  = A  + (long)batch * strideA;
  const __bf16* bt = Bt + (long)batch * strideB;
  int r0 = blockIdx.x * 16;
  int c0 = blockIdx.y * 128 + (int)(threadIdx.x >> 5) * 16;

  const __bf16* arow = a  + (long)r0 * K;
  const __bf16* bcol = bt + (long)c0 * K;

  // stage mapping: thread t copies 8 bytes (4 bf16): row t/16, k-chunk (t%16)*4
  int t  = threadIdx.x;
  int sr = t >> 4;
  int sk = (t & 15) * 4;
  unsigned lds0 = lds_addr32(&sA[0][0]) + (unsigned)(sr * 64 + sk) * 2u;
  unsigned lds1 = lds_addr32(&sA[1][0]) + (unsigned)(sr * 64 + sk) * 2u;
  const __bf16* gA = arow + (long)sr * K + sk;

  v8f acc = {0.f, 0.f, 0.f, 0.f, 0.f, 0.f, 0.f, 0.f};

  async_g2l_b64(lds0, gA);  // prologue: stage k=[0,64) into buf0
  int nIter = K / 64;
  for (int it = 0; it < nIter; ++it) {
    int kb = it * 64;
    wait_async0();
    __syncthreads();
    if (it + 1 < nIter)
      async_g2l_b64((it & 1) ? lds0 : lds1, gA + kb + 64);
    __builtin_prefetch(bcol + kb + 128, 0, 0);

    const __bf16* abuf = (it & 1) ? &sA[1][0] : &sA[0][0];
    v16bf a0 = load_a_frag(abuf, 64);
    v16bf a1 = load_a_frag(abuf + 32, 64);
    v16bf b0 = load_b_frag(bcol, K, kb);
    v16bf b1 = load_b_frag(bcol, K, kb + 32);
    acc = __builtin_amdgcn_wmma_f32_16x16x32_bf16(false, a0, false, b0,
                                                  (short)0, acc, false, false);
    acc = __builtin_amdgcn_wmma_f32_16x16x32_bf16(false, a1, false, b1,
                                                  (short)0, acc, false, false);
    __syncthreads();
  }

  int lane = threadIdx.x & 31;
  int col  = lane & 15;
  int rb   = (lane < 16) ? 0 : 8;
  if (mode == 0) {
    __bf16* Cb = (__bf16*)Cv + (long)batch * strideC;
#pragma unroll
    for (int v2 = 0; v2 < 8; ++v2)
      Cb[(long)(r0 + rb + v2) * ldc + c0 + col] = (__bf16)acc[v2];
  } else if (mode == 1) {
    __bf16* Cb = (__bf16*)Cv + (long)batch * strideC;
#pragma unroll
    for (int v2 = 0; v2 < 8; ++v2)
      Cb[(long)(c0 + col) * ldc + (r0 + rb + v2)] = (__bf16)acc[v2];
  } else {
    float* Cf = (float*)Cv + (long)batch * strideC;
#pragma unroll
    for (int v2 = 0; v2 < 8; ++v2)
      Cf[(long)(r0 + rb + v2) * ldc + c0 + col] = acc[v2];
  }
}

// Flash-style attention for one (head, 16-query-row tile).
// Q,K in (t,h) bf16 rm; V transposed (h,t) bf16 rm. Output -> Rcat (t, nh*hd) bf16.
// The 32x128 K tile (8KB, contiguous) is async-staged to LDS by all 256 threads,
// double buffered so staging s-step i+1 overlaps softmax/PV of step i.
__global__ __launch_bounds__(256) void flash_attn_wmma(
    const __bf16* __restrict__ Qb, const __bf16* __restrict__ Kb,
    const __bf16* __restrict__ Vt, __bf16* __restrict__ Rcat) {
  int head = blockIdx.y;
  int q0   = blockIdx.x * 16;
  int w    = threadIdx.x >> 5;
  int lane = threadIdx.x & 31;
  int rowoff = (lane < 16) ? 0 : 8;
  int coln   = lane & 15;

  const __bf16* Qh = Qb + (long)head * T * HD;
  const __bf16* Kh = Kb + (long)head * T * HD;
  const __bf16* Vh = Vt + (long)head * HD * T;

  __shared__ __bf16 sK[2][32 * 128];  // 2 x 8KB K-tile double buffer
  __shared__ __bf16 sP[16 * 32];
  __shared__ float  sScale[16];
  __shared__ float  sL[16];

  v8f O = {0.f, 0.f, 0.f, 0.f, 0.f, 0.f, 0.f, 0.f};
  float m[8], l[8];
  v16bf Aq[4];
  if (w == 0) {
#pragma unroll
    for (int kc = 0; kc < 4; ++kc)
      Aq[kc] = load_a_frag(Qh + (long)q0 * HD + kc * 32, HD);
#pragma unroll
    for (int v2 = 0; v2 < 8; ++v2) { m[v2] = -1e30f; l[v2] = 0.f; }
  }
  const float scale = 0.08838834764831845f; // 1/sqrt(128)

  // K-tile staging: thread t copies 32 bytes (two b128) of the contiguous 8KB tile
  int t = threadIdx.x;
  unsigned kl0 = lds_addr32(&sK[0][0]) + (unsigned)t * 32u;
  unsigned kl1 = lds_addr32(&sK[1][0]) + (unsigned)t * 32u;
  const __bf16* gK = Kh + t * 16;

  async_g2l_b128(kl0, gK);
  async_g2l_b128(kl0 + 16, gK + 8);

  const int nIter = T / 32;
  for (int it = 0; it < nIter; ++it) {
    int s0 = it * 32;
    wait_async0();
    __syncthreads();                       // K tile buf[it&1] visible; prev sP fully consumed
    if (it + 1 < nIter) {
      const __bf16* gKn = gK + (long)(s0 + 32) * HD;
      unsigned dst = (it & 1) ? kl0 : kl1;
      async_g2l_b128(dst, gKn);
      async_g2l_b128(dst + 16, gKn + 8);
    }
    const __bf16* kbuf = (it & 1) ? &sK[1][0] : &sK[0][0];

    if (w == 0) {
      v8f c0 = {0.f, 0.f, 0.f, 0.f, 0.f, 0.f, 0.f, 0.f};
      v8f c1 = {0.f, 0.f, 0.f, 0.f, 0.f, 0.f, 0.f, 0.f};
#pragma unroll
      for (int kc = 0; kc < 4; ++kc) {
        v16bf b0 = load_b_frag(kbuf, HD, kc * 32);
        v16bf b1 = load_b_frag(kbuf + 16 * HD, HD, kc * 32);
        c0 = __builtin_amdgcn_wmma_f32_16x16x32_bf16(false, Aq[kc], false, b0,
                                                     (short)0, c0, false, false);
        c1 = __builtin_amdgcn_wmma_f32_16x16x32_bf16(false, Aq[kc], false, b1,
                                                     (short)0, c1, false, false);
      }
#pragma unroll
      for (int v2 = 0; v2 < 8; ++v2) {
        float a0 = c0[v2] * scale;
        float a1 = c1[v2] * scale;
        float tm = fmaxf(a0, a1);          // per-row max across the row's 16 lanes
#pragma unroll
        for (int off = 1; off < 16; off <<= 1)
          tm = fmaxf(tm, __shfl_xor(tm, off, 32));
        float mnew = fmaxf(m[v2], tm);
        float sf = __expf(m[v2] - mnew);
        float p0 = __expf(a0 - mnew);
        float p1 = __expf(a1 - mnew);
        float rs = p0 + p1;
#pragma unroll
        for (int off = 1; off < 16; off <<= 1)
          rs += __shfl_xor(rs, off, 32);
        l[v2] = l[v2] * sf + rs;
        m[v2] = mnew;
        int row = rowoff + v2;
        sP[row * 32 + coln]      = (__bf16)p0;
        sP[row * 32 + coln + 16] = (__bf16)p1;
        if (coln == 0) sScale[row] = sf;
      }
    }
    __syncthreads();
    {
#pragma unroll
      for (int v2 = 0; v2 < 8; ++v2)
        O[v2] *= sScale[rowoff + v2];
      v16bf ap = load_a_frag(sP, 32);
      v16bf bv = load_b_frag(Vh + (long)(w * 16) * T, T, s0);
      O = __builtin_amdgcn_wmma_f32_16x16x32_bf16(false, ap, false, bv,
                                                  (short)0, O, false, false);
    }
  }

  __syncthreads();
  if (w == 0 && coln == 0) {
#pragma unroll
    for (int v2 = 0; v2 < 8; ++v2) sL[rowoff + v2] = l[v2];
  }
  __syncthreads();
#pragma unroll
  for (int v2 = 0; v2 < 8; ++v2) {
    float oo = O[v2] / sL[rowoff + v2];
    Rcat[(long)(q0 + rowoff + v2) * D + head * HD + w * 16 + coln] = (__bf16)oo;
  }
}

extern "C" void kernel_launch(void* const* d_in, const int* in_sizes, int n_in,
                              void* d_out, int out_size, void* d_ws, size_t ws_size,
                              hipStream_t stream) {
  (void)in_sizes; (void)n_in; (void)out_size; (void)ws_size;
  const float* X   = (const float*)d_in[0];
  const float* W_Q = (const float*)d_in[1];
  const float* W_K = (const float*)d_in[2];
  const float* W_V = (const float*)d_in[3];
  const float* W_O = (const float*)d_in[4];

  char* ws = (char*)d_ws;
  size_t off = 0;
  auto alloc = [&](size_t bytes) -> void* {
    void* p = ws + off;
    off += (bytes + 255) & ~(size_t)255;
    return p;
  };
  __bf16* Xb  = (__bf16*)alloc((size_t)T * D * 2);
  __bf16* Qb  = (__bf16*)alloc((size_t)NH * T * HD * 2);
  __bf16* Kb  = (__bf16*)alloc((size_t)NH * T * HD * 2);
  __bf16* Vt  = (__bf16*)alloc((size_t)NH * HD * T * 2);
  __bf16* Rc  = (__bf16*)alloc((size_t)T * D * 2);
  __bf16* Wtq = (__bf16*)alloc((size_t)NH * HD * D * 2);
  __bf16* Wtk = (__bf16*)alloc((size_t)NH * HD * D * 2);
  __bf16* Wtv = (__bf16*)alloc((size_t)NH * HD * D * 2);
  __bf16* Wto = (__bf16*)alloc((size_t)D * D * 2);

  // f32 -> bf16 conversions (weights transposed so WMMA B-frags are contiguous)
  cvt_f32_bf16<<<dim3((T * D) / 256), 256, 0, stream>>>(X, Xb, T * D);
  transpose_f32_bf16<<<dim3((D * HD) / 256, 1, NH), 256, 0, stream>>>(W_Q, Wtq, D, HD);
  transpose_f32_bf16<<<dim3((D * HD) / 256, 1, NH), 256, 0, stream>>>(W_K, Wtk, D, HD);
  transpose_f32_bf16<<<dim3((D * HD) / 256, 1, NH), 256, 0, stream>>>(W_V, Wtv, D, HD);
  transpose_f32_bf16<<<dim3((D * D) / 256, 1, 1), 256, 0, stream>>>(W_O, Wto, D, D);

  // Projections: Q,K as (t,h); V stored transposed (h,t)
  gemm_bf16_wmma<<<dim3(T / 16, 1, NH), 256, 0, stream>>>(
      Xb, Wtq, Qb, T, HD, D, HD, 0, 0, (long)HD * D, (long)T * HD);
  gemm_bf16_wmma<<<dim3(T / 16, 1, NH), 256, 0, stream>>>(
      Xb, Wtk, Kb, T, HD, D, HD, 0, 0, (long)HD * D, (long)T * HD);
  gemm_bf16_wmma<<<dim3(T / 16, 1, NH), 256, 0, stream>>>(
      Xb, Wtv, Vt, T, HD, D, T, 1, 0, (long)HD * D, (long)HD * T);

  // Streaming-softmax attention, output already concatenated (t, nh*hd)
  flash_attn_wmma<<<dim3(T / 16, NH), 256, 0, stream>>>(Qb, Kb, Vt, Rc);

  // Output projection -> fp32 d_out
  gemm_bf16_wmma<<<dim3(T / 16, D / 128, 1), 256, 0, stream>>>(
      Rc, Wto, d_out, T, D, D, D, 2, 0, 0, 0);
}
